// Network_34660386078854
// MI455X (gfx1250) — matrix operand order repaired and checked by reading, not compiled, not run on hardware
//
#include <hip/hip_runtime.h>

// ---------------------------------------------------------------------------
// CDNA5 / gfx1250 implementation. wave32 only. All dense contractions use
// v_wmma_f32_16x16x32_f16 (f32 data downconverted per-fragment, f32 accum).
// Fragment layouts in gemm/conv (flash loads A and B in ISA table order):
//   A (16x32 f16): lane row = L&15; k = half*16 + e
//   B (32x16 f16): lane col = L&15; k = half*16 + e
//   C (16x16 f32): lane col = L&15; element r -> row = r + half*8
// Loads are branch-free: indices are clamped (garbage only feeds unstored C
// rows/cols) and K-tail masking is done with v_cndmask on loaded values.
// ---------------------------------------------------------------------------

typedef __attribute__((ext_vector_type(16))) _Float16 v16h;
typedef __attribute__((ext_vector_type(8)))  float    v8f;

#define WMMA_F32_F16(a, b, c) \
  __builtin_amdgcn_wmma_f32_16x16x32_f16(false, (a), false, (b), (short)0, (c), false, false)

__device__ __forceinline__ int imin(int a, int b) { return a < b ? a : b; }
__device__ __forceinline__ int imax(int a, int b) { return a > b ? a : b; }

// ---------------------------------------------------------------------------
// Generic batched GEMM: C = act(A[M,K] * B[K,N] + bias[N]); row-major, strided
// batch. One wave computes one 16x16 C tile.
// ---------------------------------------------------------------------------
__global__ __launch_bounds__(32)
void gemm_kernel(const float* __restrict__ A, const float* __restrict__ Bm,
                 const float* __restrict__ bias, float* __restrict__ C,
                 int M, int N, int K, int lda, int ldb, int ldc, int relu,
                 long sA, long sB, long sC, long sBias)
{
  const int lane = threadIdx.x & 31;
  const int half = lane >> 4;
  const int nn   = lane & 15;
  const long bz  = blockIdx.z;
  A  += bz * sA;
  Bm += bz * sB;
  C  += bz * sC;
  if (bias) bias += bz * sBias;

  const int mClamp = imin(blockIdx.y * 16 + nn, M - 1);
  const int nClamp = imin(blockIdx.x * 16 + nn, N - 1);
  const float* __restrict__ Arow = A + (long)mClamp * lda;
  const float* __restrict__ Bcol = Bm + nClamp;

  const int kA0 = half * 16;       // A run 0: k = kA0..kA0+7
  const int kA1 = half * 16 + 8;   // A run 1
  const int kB0 = half * 16;       // B: k = kB0..kB0+15

  v8f acc = {};
  int k0 = 0;
  const bool vec = ((lda & 3) == 0) && ((((unsigned long long)A) & 15) == 0);
  if (vec) {
    for (; k0 + 32 <= K; k0 += 32) {
      v16h af, bf;
      const float4* a0 = (const float4*)(Arow + k0 + kA0);
      const float4* a1 = (const float4*)(Arow + k0 + kA1);
      float4 p0 = a0[0], p1 = a0[1], p2 = a1[0], p3 = a1[1];
      af[0]  = (_Float16)p0.x; af[1]  = (_Float16)p0.y; af[2]  = (_Float16)p0.z; af[3]  = (_Float16)p0.w;
      af[4]  = (_Float16)p1.x; af[5]  = (_Float16)p1.y; af[6]  = (_Float16)p1.z; af[7]  = (_Float16)p1.w;
      af[8]  = (_Float16)p2.x; af[9]  = (_Float16)p2.y; af[10] = (_Float16)p2.z; af[11] = (_Float16)p2.w;
      af[12] = (_Float16)p3.x; af[13] = (_Float16)p3.y; af[14] = (_Float16)p3.z; af[15] = (_Float16)p3.w;
      const float* bp = Bcol + (long)(k0 + kB0) * ldb;
#pragma unroll
      for (int e = 0; e < 16; ++e) bf[e] = (_Float16)bp[(long)e * ldb];
      __builtin_prefetch(Arow + k0 + 32, 0, 3);
      acc = WMMA_F32_F16(af, bf, acc);
    }
  } else {
    for (; k0 + 32 <= K; k0 += 32) {
      v16h af, bf;
#pragma unroll
      for (int e = 0; e < 8; ++e) {
        af[e]     = (_Float16)Arow[k0 + kA0 + e];
        af[e + 8] = (_Float16)Arow[k0 + kA1 + e];
      }
      const float* bp = Bcol + (long)(k0 + kB0) * ldb;
#pragma unroll
      for (int e = 0; e < 16; ++e) bf[e] = (_Float16)bp[(long)e * ldb];
      acc = WMMA_F32_F16(af, bf, acc);
    }
  }
  if (k0 < K) {  // K tail: clamped loads + value masks (cndmask, no branches)
    v16h af, bf;
#pragma unroll
    for (int e = 0; e < 8; ++e) {
      int ka = k0 + kA0 + e;
      float v0 = Arow[imin(ka, K - 1)];
      af[e] = (_Float16)(ka < K ? v0 : 0.0f);
      int kb = k0 + kA1 + e;
      float v1 = Arow[imin(kb, K - 1)];
      af[e + 8] = (_Float16)(kb < K ? v1 : 0.0f);
    }
#pragma unroll
    for (int e = 0; e < 16; ++e) {
      int kk = k0 + kB0 + e;
      float v = Bcol[(long)imin(kk, K - 1) * ldb];
      bf[e] = (_Float16)(kk < K ? v : 0.0f);
    }
    acc = WMMA_F32_F16(af, bf, acc);
  }

  const float bval = bias ? bias[nClamp] : 0.0f;
#pragma unroll
  for (int r = 0; r < 8; ++r) {
    int m = blockIdx.y * 16 + r + half * 8;
    int n = blockIdx.x * 16 + nn;
    if (m < M && n < N) {
      float v = acc[r] + bval;
      if (relu) v = fmaxf(v, 0.0f);
      C[(long)m * ldc + n] = v;
    }
  }
}

// ---------------------------------------------------------------------------
// Implicit-GEMM conv / lhs-dilated deconv (NCHW / OIHW), templated on kernel
// size and lhs-dilation so index decode divides by constants.
// A = W[Cout, Cin*KSZ*KSZ]; B = gathered patches [Cin*KSZ*KSZ, pixels].
// DIL==2 implements lhs_dilation=2 transpose-convs.
// ---------------------------------------------------------------------------
template <int KSZ, int DIL>
__global__ __launch_bounds__(32)
void conv_kernel(const float* __restrict__ X, const float* __restrict__ W,
                 const float* __restrict__ bias, float* __restrict__ Y,
                 int Bn, int Cin, int Hin, int Win, int Cout,
                 int Hout, int Wout, int stride, int pad, int relu)
{
  constexpr int KK2 = KSZ * KSZ;
  const int lane = threadIdx.x & 31;
  const int half = lane >> 4;
  const int nn   = lane & 15;
  const int Kdim = Cin * KK2;
  const int P    = Bn * Hout * Wout;

  const int coC = imin(blockIdx.y * 16 + nn, Cout - 1);
  const int p_b = blockIdx.x * 16 + nn;
  const int pC  = imin(p_b, P - 1);
  const int b   = pC / (Hout * Wout);
  const int rem = pC - b * (Hout * Wout);
  const int oh  = rem / Wout;
  const int ow  = rem - oh * Wout;
  const int ohs = oh * stride - pad;
  const int ows = ow * stride - pad;
  const float* __restrict__ Wrow = W + (long)coC * Kdim;
  const float* __restrict__ Xb   = X + (long)b * Cin * Hin * Win;

  const int kA0 = half * 16;
  const int kA1 = half * 16 + 8;
  const int kB0 = half * 16;

  v8f acc = {};
  int k0 = 0;
  for (; k0 + 32 <= Kdim; k0 += 32) {
    v16h af, bf;
#pragma unroll
    for (int e = 0; e < 8; ++e) {
      af[e]     = (_Float16)Wrow[k0 + kA0 + e];
      af[e + 8] = (_Float16)Wrow[k0 + kA1 + e];
    }
#pragma unroll
    for (int e = 0; e < 16; ++e) {
      int kb = k0 + kB0 + e;
      int ci = kb / KK2;
      int r2 = kb - ci * KK2;
      int kh = r2 / KSZ;
      int kw = r2 - kh * KSZ;
      int th = ohs + kh, tw = ows + kw;
      int ih, iw; bool ok;
      if (DIL == 1) {
        ih = th; iw = tw;
        ok = (th >= 0) & (tw >= 0) & (th < Hin) & (tw < Win);
      } else {
        ih = th >> 1; iw = tw >> 1;
        ok = (th >= 0) & (tw >= 0) & ((th & 1) == 0) & ((tw & 1) == 0) &
             (ih < Hin) & (iw < Win);
      }
      int ihC = imin(imax(ih, 0), Hin - 1);
      int iwC = imin(imax(iw, 0), Win - 1);
      float v = Xb[((long)ci * Hin + ihC) * Win + iwC];
      bf[e] = (_Float16)(ok ? v : 0.0f);
    }
    acc = WMMA_F32_F16(af, bf, acc);
  }
  if (k0 < Kdim) {
    v16h af, bf;
#pragma unroll
    for (int e = 0; e < 8; ++e) {
      int ka = k0 + kA0 + e;
      float v0 = Wrow[imin(ka, Kdim - 1)];
      af[e] = (_Float16)(ka < Kdim ? v0 : 0.0f);
      int kb = k0 + kA1 + e;
      float v1 = Wrow[imin(kb, Kdim - 1)];
      af[e + 8] = (_Float16)(kb < Kdim ? v1 : 0.0f);
    }
#pragma unroll
    for (int e = 0; e < 16; ++e) {
      int kbu = k0 + kB0 + e;
      int kb  = imin(kbu, Kdim - 1);
      int ci = kb / KK2;
      int r2 = kb - ci * KK2;
      int kh = r2 / KSZ;
      int kw = r2 - kh * KSZ;
      int th = ohs + kh, tw = ows + kw;
      int ih, iw; bool ok;
      if (DIL == 1) {
        ih = th; iw = tw;
        ok = (th >= 0) & (tw >= 0) & (th < Hin) & (tw < Win);
      } else {
        ih = th >> 1; iw = tw >> 1;
        ok = (th >= 0) & (tw >= 0) & ((th & 1) == 0) & ((tw & 1) == 0) &
             (ih < Hin) & (iw < Win);
      }
      ok = ok & (kbu < Kdim);
      int ihC = imin(imax(ih, 0), Hin - 1);
      int iwC = imin(imax(iw, 0), Win - 1);
      float v = Xb[((long)ci * Hin + ihC) * Win + iwC];
      bf[e] = (_Float16)(ok ? v : 0.0f);
    }
    acc = WMMA_F32_F16(af, bf, acc);
  }
#pragma unroll
  for (int r = 0; r < 8; ++r) {
    int co = blockIdx.y * 16 + r + half * 8;
    if (co < Cout && p_b < P) {
      float v = acc[r] + (bias ? bias[co] : 0.0f);
      if (relu) v = fmaxf(v, 0.0f);
      Y[(((long)b * Cout + co) * Hout + oh) * Wout + ow] = v;
    }
  }
}

// ---------------------------------------------------------------------------
// Flash-style attention, one wave per (q-tile-of-16, head, batch).
// Heads packed in feature dim: head h owns cols [h*DH, (h+1)*DH). D = 4*DH.
// All loads clamped/branch-free; masked keys get -1e38 before softmax, and
// the zero columns of the padded 16x32 P fragment nullify V rows 16..31.
// ---------------------------------------------------------------------------
template <int DH>
__global__ __launch_bounds__(32)
void flash_kernel(const float* __restrict__ Q, const float* __restrict__ K,
                  const float* __restrict__ V, float* __restrict__ O,
                  int Lq, int Lk)
{
  constexpr int D  = DH * 4;
  constexpr int NC = (DH + 31) / 32;  // K-chunks of 32 over head dim
  constexpr int NT = DH / 16;         // 16-col output tiles over head dim
  const int lane = threadIdx.x & 31;
  const int half = lane >> 4;
  const int nn   = lane & 15;
  const int q0   = blockIdx.x * 16;
  const int h    = blockIdx.y;
  const long b   = blockIdx.z;
  const float scale = rsqrtf((float)DH);

  __shared__ float s_p[16][17];
  __shared__ float s_m[16], s_l[16], s_al[16];
  if (lane < 16) { s_m[lane] = -3.0e38f; s_l[lane] = 0.0f; }
  __syncthreads();

  // Q fragments: two contiguous 8-float runs per 32-chunk -> float4 loads.
  const int qrC = imin(q0 + nn, Lq - 1);
  const float* __restrict__ Qrow = Q + ((b * Lq) + qrC) * D + h * DH;
  v16h qf[NC];
#pragma unroll
  for (int c = 0; c < NC; ++c) {
    {
      const float4* qp = (const float4*)(Qrow + c * 32 + half * 8);
      float4 p0 = qp[0], p1 = qp[1];
      qf[c][0] = (_Float16)p0.x; qf[c][1] = (_Float16)p0.y;
      qf[c][2] = (_Float16)p0.z; qf[c][3] = (_Float16)p0.w;
      qf[c][4] = (_Float16)p1.x; qf[c][5] = (_Float16)p1.y;
      qf[c][6] = (_Float16)p1.z; qf[c][7] = (_Float16)p1.w;
    }
    if constexpr (DH >= 32) {
      const float4* qp = (const float4*)(Qrow + c * 32 + 16 + half * 8);
      float4 p0 = qp[0], p1 = qp[1];
      qf[c][8]  = (_Float16)p0.x; qf[c][9]  = (_Float16)p0.y;
      qf[c][10] = (_Float16)p0.z; qf[c][11] = (_Float16)p0.w;
      qf[c][12] = (_Float16)p1.x; qf[c][13] = (_Float16)p1.y;
      qf[c][14] = (_Float16)p1.z; qf[c][15] = (_Float16)p1.w;
    } else {
#pragma unroll
      for (int e = 8; e < 16; ++e) qf[c][e] = (_Float16)0.0f;  // f >= DH pad
    }
  }

  v8f oacc[NT];
#pragma unroll
  for (int t = 0; t < NT; ++t) oacc[t] = (v8f){};

  const int nkt = (Lk + 15) / 16;
  for (int kt = 0; kt < nkt; ++kt) {
    const int krC = imin(kt * 16 + nn, Lk - 1);
    // S tile = Q * K^T ; K fragment is two 8-float runs per lane (table order).
    v8f sacc = {};
#pragma unroll
    for (int c = 0; c < NC; ++c) {
      v16h kf;
      const float4* kp = (const float4*)(K + ((b * Lk) + krC) * D + h * DH +
                                         c * 32 + half * 8);
      const float4* kq = (const float4*)((const float*)kp + ((DH >= 32) ? 16 : 0));
      float4 p0 = kp[0], p1 = kp[1], p2 = kq[0], p3 = kq[1];
      kf[0]  = (_Float16)p0.x; kf[1]  = (_Float16)p0.y; kf[2]  = (_Float16)p0.z; kf[3]  = (_Float16)p0.w;
      kf[4]  = (_Float16)p1.x; kf[5]  = (_Float16)p1.y; kf[6]  = (_Float16)p1.z; kf[7]  = (_Float16)p1.w;
      kf[8]  = (_Float16)p2.x; kf[9]  = (_Float16)p2.y; kf[10] = (_Float16)p2.z; kf[11] = (_Float16)p2.w;
      kf[12] = (_Float16)p3.x; kf[13] = (_Float16)p3.y; kf[14] = (_Float16)p3.z; kf[15] = (_Float16)p3.w;
      if constexpr (DH < 32) {               // f >= DH (elements 8..15) pad
        for (int e = 8; e < 16; ++e) kf[e] = (_Float16)0.0f;
      }
      sacc = WMMA_F32_F16(qf[c], kf, sacc);
    }
#pragma unroll
    for (int r = 0; r < 8; ++r) {
      int kr = kt * 16 + nn;
      s_p[r + half * 8][nn] = (kr < Lk) ? sacc[r] * scale : -1.0e38f;
    }
    __syncthreads();

    // online softmax: lane i owns query row i
    if (lane < 16) {
      float mo = s_m[lane], rm = mo;
      for (int j = 0; j < 16; ++j) rm = fmaxf(rm, s_p[lane][j]);
      float alpha = __expf(mo - rm);
      float ls = 0.0f;
      for (int j = 0; j < 16; ++j) {
        float p = __expf(s_p[lane][j] - rm);
        s_p[lane][j] = p;
        ls += p;
      }
      s_l[lane]  = alpha * s_l[lane] + ls;
      s_m[lane]  = rm;
      s_al[lane] = alpha;
    }
    __syncthreads();

#pragma unroll
    for (int t = 0; t < NT; ++t)
#pragma unroll
      for (int r = 0; r < 8; ++r)
        oacc[t][r] = oacc[t][r] * s_al[r + half * 8];

    // P fragment (16x32, cols 16..31 zero)
    v16h pf;
#pragma unroll
    for (int e = 0; e < 8; ++e) {
      pf[e]     = (_Float16)s_p[nn][half * 8 + e];
      pf[e + 8] = (_Float16)0.0f;
    }
    // O += P * V  (V rows 16..31 are don't-care: P cols 16..31 are zero)
#pragma unroll
    for (int t = 0; t < NT; ++t) {
      v16h vf;
#pragma unroll
      for (int e = 0; e < 16; ++e) {
        int kr = imin(kt * 16 + half * 8 + (e < 8 ? e : e + 8), Lk - 1);
        vf[e] = (_Float16)V[((b * Lk) + kr) * D + h * DH + t * 16 + nn];
      }
      oacc[t] = WMMA_F32_F16(pf, vf, oacc[t]);
    }
    __syncthreads();
  }

#pragma unroll
  for (int t = 0; t < NT; ++t)
#pragma unroll
    for (int r = 0; r < 8; ++r) {
      int row = q0 + r + half * 8;
      if (row < Lq) {
        float l = s_l[r + half * 8];
        O[((b * Lq) + row) * D + h * DH + t * 16 + nn] = oacc[t][r] / l;
      }
    }
}

// ---------------------------------------------------------------------------
// Small helper kernels: transpose, maxpool, softmax, add+LN (async global->LDS
// staging, ASYNCcnt), pred-offset combine, farthest point sampling.
// ---------------------------------------------------------------------------
__global__ void transpose_kernel(const float* __restrict__ in, float* __restrict__ out,
                                 int Bn, int C, int P)
{
  long i = (long)blockIdx.x * blockDim.x + threadIdx.x;
  long tot = (long)Bn * C * P;
  if (i >= tot) return;
  int b = (int)(i / ((long)C * P));
  long r = i % ((long)C * P);
  int c = (int)(r / P), p = (int)(r % P);
  out[((long)b * P + p) * C + c] = in[i];
}

__global__ void maxpool_kernel(const float* __restrict__ x, float* __restrict__ out,
                               int N, int C)
{
  int b = blockIdx.x, c = threadIdx.x;
  float m = -3.0e38f;
  for (int n = 0; n < N; ++n) m = fmaxf(m, x[((long)b * N + n) * C + c]);
  out[(long)b * C + c] = m;
}

__global__ void softmax_kernel(const float* __restrict__ in, float* __restrict__ out, int cols)
{
  int row = blockIdx.x, tid = threadIdx.x;
  __shared__ float red[128];
  float v = in[(long)row * cols + tid];
  red[tid] = v; __syncthreads();
  for (int o = 64; o > 0; o >>= 1) {
    if (tid < o) red[tid] = fmaxf(red[tid], red[tid + o]);
    __syncthreads();
  }
  float m = red[0]; __syncthreads();
  float e = __expf(v - m);
  red[tid] = e; __syncthreads();
  for (int o = 64; o > 0; o >>= 1) {
    if (tid < o) red[tid] += red[tid + o];
    __syncthreads();
  }
  out[(long)row * cols + tid] = e / red[0];
}

__global__ __launch_bounds__(64)
void add_ln_kernel(const float* __restrict__ x, float* __restrict__ resid,
                   const float* __restrict__ g, const float* __restrict__ bta, int d)
{
  const long row = blockIdx.x;
  const int tid = threadIdx.x;
  __shared__ float s_buf[512];   // [0..255] = x row, [256..511] = resid row
  __shared__ float red[64];
  const int cnt = d / 64;
  // Stage both rows into LDS with CDNA5 async global->LDS copies (ASYNCcnt).
  for (int i = 0; i < cnt; ++i) {
    int c = tid + i * 64;
    unsigned lds0 = (unsigned)(unsigned long long)&s_buf[c];
    unsigned lds1 = (unsigned)(unsigned long long)&s_buf[256 + c];
    unsigned long long ga0 = (unsigned long long)(const void*)&x[row * d + c];
    unsigned long long ga1 = (unsigned long long)(const void*)&resid[row * d + c];
    asm volatile("global_load_async_to_lds_b32 %0, %1, off"
                 :: "v"(lds0), "v"(ga0) : "memory");
    asm volatile("global_load_async_to_lds_b32 %0, %1, off"
                 :: "v"(lds1), "v"(ga1) : "memory");
  }
  asm volatile("s_wait_asynccnt 0x0" ::: "memory");
  __syncthreads();

  float vals[4];
  float s = 0.0f;
  for (int i = 0; i < cnt; ++i) {
    int c = tid + i * 64;
    float v = s_buf[c] + s_buf[256 + c];
    vals[i] = v; s += v;
  }
  red[tid] = s; __syncthreads();
  for (int o = 32; o > 0; o >>= 1) { if (tid < o) red[tid] += red[tid + o]; __syncthreads(); }
  float mean = red[0] / d; __syncthreads();
  float s2 = 0.0f;
  for (int i = 0; i < cnt; ++i) { float t = vals[i] - mean; s2 += t * t; }
  red[tid] = s2; __syncthreads();
  for (int o = 32; o > 0; o >>= 1) { if (tid < o) red[tid] += red[tid + o]; __syncthreads(); }
  float rs = rsqrtf(red[0] / d + 1e-5f);
  for (int i = 0; i < cnt; ++i) {
    int c = tid + i * 64;
    resid[row * d + c] = (vals[i] - mean) * rs * g[c] + bta[c];
  }
}

__global__ void pred_kernel(const float* __restrict__ fin, const float* __restrict__ off,
                            float* __restrict__ pred, long tot)
{
  long i = (long)blockIdx.x * blockDim.x + threadIdx.x;
  if (i >= tot) return;
  pred[i] = fin[(i / 6) * 3 + (i % 3)] + off[i];
}

__global__ __launch_bounds__(256)
void fps_kernel(const float* __restrict__ pts, float* __restrict__ out,
                int N, int m, long sIn, long sOut)
{
  const int b = blockIdx.x;
  pts += (long)b * sIn;
  out += (long)b * sOut;
  const int tid = threadIdx.x;
  __shared__ float dist[4096];
  __shared__ float rmax[256];
  __shared__ int   ridx[256];
  __shared__ int   s_far;
  for (int i = tid; i < N; i += 256) dist[i] = 1e10f;
  if (tid == 0) s_far = 0;
  __syncthreads();
  for (int it = 0; it < m; ++it) {
    int far = s_far;
    float cx = pts[(long)far * 3 + 0];
    float cy = pts[(long)far * 3 + 1];
    float cz = pts[(long)far * 3 + 2];
    if (tid == 0) {
      out[(long)it * 3 + 0] = cx;
      out[(long)it * 3 + 1] = cy;
      out[(long)it * 3 + 2] = cz;
    }
    float best = -1.0f; int bidx = 0;
    for (int i = tid; i < N; i += 256) {
      float dx = pts[(long)i * 3 + 0] - cx;
      float dy = pts[(long)i * 3 + 1] - cy;
      float dz = pts[(long)i * 3 + 2] - cz;
      float nd = fminf(dist[i], dx * dx + dy * dy + dz * dz);
      dist[i] = nd;
      if (nd > best) { best = nd; bidx = i; }
    }
    rmax[tid] = best; ridx[tid] = bidx;
    __syncthreads();
    for (int o = 128; o > 0; o >>= 1) {
      if (tid < o) {
        if (rmax[tid + o] > rmax[tid] ||
            (rmax[tid + o] == rmax[tid] && ridx[tid + o] < ridx[tid])) {
          rmax[tid] = rmax[tid + o]; ridx[tid] = ridx[tid + o];
        }
      }
      __syncthreads();
    }
    if (tid == 0) s_far = ridx[0];
    __syncthreads();
  }
}

// ---------------------------------------------------------------------------
// Host orchestration. Params follow jax tree_leaves (alphabetical) order after
// x_part (d_in[0]) and view (d_in[1]). Workspace need: ~128 MB.
// Outputs concatenated: complete | final | pc_generate | pred_pcds.
// ---------------------------------------------------------------------------
extern "C" void kernel_launch(void* const* d_in, const int* in_sizes, int n_in,
                              void* d_out, int out_size, void* d_ws, size_t ws_size,
                              hipStream_t stream)
{
  (void)in_sizes; (void)n_in; (void)out_size; (void)ws_size;
  const float* x_part = (const float*)d_in[0];
  const float* view   = (const float*)d_in[1];
  auto P = [&](int i) -> const float* { return (const float*)d_in[2 + i]; };
  float* ws  = (float*)d_ws;
  float* out = (float*)d_out;

  enum {
    A00 = 0, A01 = 8, A10 = 16, A11 = 24, A20 = 32, A21 = 40,
    // per a-block: bk+0, bo+1, bq+2, bv+3, wk+4, wo+5, wq+6, wv+7
    BASIS = 48, C61_B = 49, C61_W = 50, C62_B = 51, C62_W = 52,
    D1_B = 53, D1_W = 54, D2_B = 55, D2_W = 56, D3_B = 57, D3_W = 58,
    ENC_B1 = 59, ENC_B2 = 60, ENC_B3 = 61, ENC_W1 = 62, ENC_W2 = 63, ENC_W3 = 64,
    FC0D_B = 65, FC0D_W = 66, FC0U_B = 67, FC0U_W = 68,
    FC1D_B = 69, FC1D_W = 70, FC1U_B = 71, FC1U_W = 72,
    FC2D_B = 73, FC2D_W = 74, FC2U_B = 75, FC2U_W = 76,
    GEN_B = 77, GEN_W = 78,
    LN00_B = 79, LN00_G = 80, LN01_B = 81, LN01_G = 82,
    LN10_B = 83, LN10_G = 84, LN11_B = 85, LN11_G = 86,
    LN20_B = 87, LN20_G = 88, LN21_B = 89, LN21_G = 90,
    MAP_B = 91, MAP_W = 92,
    PR_B1 = 93, PR_B2 = 94, PR_W1 = 95, PR_W2 = 96,
    R1_B = 97, R1_W = 98, R2_B = 99, R2_W = 100,
    R3_B = 101, R3_W = 102, R4_B = 103, R4_W = 104
  };

  // workspace bump arena (floats)
  size_t bo_ = 0;
  auto alloc = [&](size_t n) { float* p = ws + bo_; bo_ += n; return p; };
  float* XP      = alloc(49152);    // [8,2048,3]
  float* F1      = alloc(1048576);  // [16384,64]
  float* F2      = alloc(2097152);  // [16384,128]
  float* XFEAT   = alloc(4194304);  // [16384,256]
  float* GF      = alloc(2048);     // [8,256]
  float* HBUF    = alloc(262144);   // [8,4,128,64]
  float* SBASIS  = alloc(262144);   // [4,512,128]
  float* PTSB    = alloc(1048576);  // [8,4,512,64]
  float* X1      = alloc(49152);    // [8,2048,3]
  float* X1_1024 = alloc(24576);    // [8,1024,3]
  float* V1b     = alloc(1605632);  // [8,64,56,56]
  float* V2b     = alloc(802816);   // [8,128,28,28]
  float* V3b     = alloc(401408);   // [8,256,14,14]
  float* V4b     = alloc(200704);   // [8,512,7,7]
  float* DE1     = alloc(401408);   // [8,256,14,14]
  float* DE1T    = alloc(401408);   // [8,196,256]
  float* X256    = alloc(2097152);  // [8,1024,256]
  float* DE2     = alloc(802816);   // [8,128,28,28]
  float* DE2T    = alloc(802816);   // [8,784,128]
  float* X512    = alloc(1048576);  // [8,1024,128]
  float* DE3     = alloc(1605632);  // [8,64,56,56]
  float* DE3T    = alloc(1605632);  // [8,3136,64]
  float* X1K64   = alloc(524288);   // [8,1024,64]
  float* XCAT    = alloc(73728);    // [8,3072,3]
  float* C61O    = alloc(802816);   // [8,32,56,56]
  float* C62O    = alloc(75264);    // [8,3,56,56]
  float* HH      = alloc(1048576);  // [16384,64]
  float* OFFB    = alloc(98304);    // [16384,6]
  float* QS      = alloc(2097152);
  float* KS      = alloc(2097152);
  float* VS      = alloc(2097152);
  float* AOS     = alloc(2097152);

  float* COMPLETE = out;            // [8,2048,3]
  float* FINAL    = out + 49152;    // [8,2048,3]
  float* PCGEN    = out + 98304;    // [8,2048,3]
  float* PRED     = out + 147456;   // [8,4096,3]

  auto gemm = [&](const float* A, const float* Bm, const float* bias, float* C,
                  int M, int N, int K, int lda, int ldb, int ldc, int relu,
                  int batch, long sA, long sB, long sC, long sBias) {
    dim3 g((N + 15) / 16, (M + 15) / 16, batch);
    gemm_kernel<<<g, 32, 0, stream>>>(A, Bm, bias, C, M, N, K, lda, ldb, ldc,
                                      relu, sA, sB, sC, sBias);
  };
  auto conv = [&](const float* X, int Cin, int Hin, int Win,
                  const float* W, const float* bias, float* Y,
                  int Cout, int Kk, int Hout, int Wout,
                  int stride, int pad, int dil, int relu) {
    int Pp = 8 * Hout * Wout;
    dim3 g((Pp + 15) / 16, (Cout + 15) / 16, 1);
    if (Kk == 7 && dil == 1)
      conv_kernel<7, 1><<<g, 32, 0, stream>>>(X, W, bias, Y, 8, Cin, Hin, Win,
                                              Cout, Hout, Wout, stride, pad, relu);
    else if (Kk == 3 && dil == 1)
      conv_kernel<3, 1><<<g, 32, 0, stream>>>(X, W, bias, Y, 8, Cin, Hin, Win,
                                              Cout, Hout, Wout, stride, pad, relu);
    else if (Kk == 5)
      conv_kernel<5, 2><<<g, 32, 0, stream>>>(X, W, bias, Y, 8, Cin, Hin, Win,
                                              Cout, Hout, Wout, stride, pad, relu);
    else
      conv_kernel<3, 2><<<g, 32, 0, stream>>>(X, W, bias, Y, 8, Cin, Hin, Win,
                                              Cout, Hout, Wout, stride, pad, relu);
  };
  auto flash = [&](const float* Qp, const float* Kp, const float* Vp, float* Op,
                   int Lq, int Lk, int d) {
    dim3 g((Lq + 15) / 16, 4, 8);
    if (d == 256)      flash_kernel<64><<<g, 32, 0, stream>>>(Qp, Kp, Vp, Op, Lq, Lk);
    else if (d == 128) flash_kernel<32><<<g, 32, 0, stream>>>(Qp, Kp, Vp, Op, Lq, Lk);
    else               flash_kernel<16><<<g, 32, 0, stream>>>(Qp, Kp, Vp, Op, Lq, Lk);
  };
  auto mha = [&](float* qbuf, int Lq, float* kvbuf, int Lk, int d,
                 int base, int lnG, int lnB) {
    gemm(qbuf,  P(base + 6), P(base + 2), QS, 8 * Lq, d, d, d, d, d, 0, 1, 0, 0, 0, 0);
    gemm(kvbuf, P(base + 4), P(base + 0), KS, 8 * Lk, d, d, d, d, d, 0, 1, 0, 0, 0, 0);
    gemm(kvbuf, P(base + 7), P(base + 3), VS, 8 * Lk, d, d, d, d, d, 0, 1, 0, 0, 0, 0);
    flash(QS, KS, VS, AOS, Lq, Lk, d);
    gemm(AOS, P(base + 5), P(base + 1), KS, 8 * Lq, d, d, d, d, d, 0, 1, 0, 0, 0, 0);
    add_ln_kernel<<<8 * Lq, 64, 0, stream>>>(KS, qbuf, P(lnG), P(lnB), d);
  };

  // ---- encoder ----
  {
    long tot = 8L * 3 * 2048;
    transpose_kernel<<<(int)((tot + 255) / 256), 256, 0, stream>>>(x_part, XP, 8, 3, 2048);
  }
  gemm(XP, P(ENC_W1), P(ENC_B1), F1,    16384,  64,   3,   3,  64,  64, 1, 1, 0, 0, 0, 0);
  gemm(F1, P(ENC_W2), P(ENC_B2), F2,    16384, 128,  64,  64, 128, 128, 1, 1, 0, 0, 0, 0);
  gemm(F2, P(ENC_W3), P(ENC_B3), XFEAT, 16384, 256, 128, 128, 256, 256, 0, 1, 0, 0, 0, 0);
  maxpool_kernel<<<8, 256, 0, stream>>>(XFEAT, GF, 2048, 256);

  // ---- branch generation (AXform stand-in) ----
  for (int r = 0; r < 4; ++r)
    gemm(GF, P(MAP_W) + (long)r * 256 * 8192, P(MAP_B) + (long)r * 8192,
         HBUF + (long)r * 8192, 8, 8192, 256, 256, 8192, 32768, 0, 1, 0, 0, 0, 0);
  softmax_kernel<<<2048, 128, 0, stream>>>(P(BASIS), SBASIS, 128);
  for (int r = 0; r < 4; ++r)   // pts = attn @ h, batched over b
    gemm(SBASIS + (long)r * 65536, HBUF + (long)r * 8192, nullptr,
         PTSB + (long)r * 32768, 512, 64, 128, 128, 64, 64, 0, 8, 0, 32768, 131072, 0);
  for (int r = 0; r < 4; ++r)   // x_1 = pts @ gen_w + gen_b
    gemm(PTSB + (long)r * 32768, P(GEN_W) + (long)r * 192, P(GEN_B) + (long)r * 3,
         X1 + (long)r * 1536, 512, 3, 64, 64, 3, 3, 0, 8, 131072, 0, 6144, 0);

  fps_kernel<<<8, 256, 0, stream>>>(XP, FINAL + 3072, 2048, 1024, 6144, 6144); // x_partial
  fps_kernel<<<8, 256, 0, stream>>>(X1, X1_1024,      2048, 1024, 6144, 3072);

  // ---- ResNet stand-in ----
  conv(view, 3, 224, 224, P(R1_W), P(R1_B), V1b,  64, 7, 56, 56, 4, 3, 1, 1);
  conv(V1b, 64,  56,  56, P(R2_W), P(R2_B), V2b, 128, 3, 28, 28, 2, 1, 1, 1);
  conv(V2b, 128, 28,  28, P(R3_W), P(R3_B), V3b, 256, 3, 14, 14, 2, 1, 1, 1);
  conv(V3b, 256, 14,  14, P(R4_W), P(R4_B), V4b, 512, 3,  7,  7, 2, 1, 1, 1);

  // ---- level 0 ----
  conv(V4b, 512, 7, 7, P(D1_W), P(D1_B), DE1, 256, 5, 14, 14, 1, 2, 2, 0);
  {
    long tot = 8L * 256 * 196;
    transpose_kernel<<<(int)((tot + 255) / 256), 256, 0, stream>>>(DE1, DE1T, 8, 256, 196);
  }
  gemm(X1_1024, P(FC0U_W), P(FC0U_B), X256, 8192, 256, 3, 3, 256, 256, 0, 1, 0, 0, 0, 0);
  mha(DE1T, 196, X256, 1024, 256, A00, LN00_G, LN00_B);
  mha(X256, 1024, DE1T, 196, 256, A01, LN01_G, LN01_B);
  gemm(X256, P(FC0D_W), P(FC0D_B), XCAT, 1024, 3, 256, 256, 3, 3, 0, 8, 262144, 0, 9216, 0);

  // ---- level 1 (DE1T raw-reshaped to [8,256,14,14]) ----
  conv(DE1T, 256, 14, 14, P(D2_W), P(D2_B), DE2, 128, 5, 28, 28, 1, 2, 2, 0);
  {
    long tot = 8L * 128 * 784;
    transpose_kernel<<<(int)((tot + 255) / 256), 256, 0, stream>>>(DE2, DE2T, 8, 128, 784);
  }
  gemm(X1_1024, P(FC1U_W), P(FC1U_B), X512, 8192, 128, 3, 3, 128, 128, 0, 1, 0, 0, 0, 0);
  mha(DE2T, 784, X512, 1024, 128, A10, LN10_G, LN10_B);
  mha(X512, 1024, DE2T, 784, 128, A11, LN11_G, LN11_B);
  gemm(X512, P(FC1D_W), P(FC1D_B), XCAT + 3072, 1024, 3, 128, 128, 3, 3, 0, 8, 131072, 0, 9216, 0);

  // ---- level 2 (DE2T raw-reshaped to [8,128,28,28]) ----
  conv(DE2T, 128, 28, 28, P(D3_W), P(D3_B), DE3, 64, 3, 56, 56, 1, 1, 2, 0);
  {
    long tot = 8L * 64 * 3136;
    transpose_kernel<<<(int)((tot + 255) / 256), 256, 0, stream>>>(DE3, DE3T, 8, 64, 3136);
  }
  gemm(X1_1024, P(FC2U_W), P(FC2U_B), X1K64, 8192, 64, 3, 3, 64, 64, 0, 1, 0, 0, 0, 0);
  mha(DE3T, 3136, X1K64, 1024, 64, A20, LN20_G, LN20_B);
  mha(X1K64, 1024, DE3T, 3136, 64, A21, LN21_G, LN21_B);
  gemm(X1K64, P(FC2D_W), P(FC2D_B), XCAT + 6144, 1024, 3, 64, 64, 3, 3, 0, 8, 65536, 0, 9216, 0);

  // ---- fuse branches, image head ----
  fps_kernel<<<8, 256, 0, stream>>>(XCAT, FINAL, 3072, 1024, 9216, 6144);
  conv(DE3T, 64, 56, 56, P(C61_W), P(C61_B), C61O, 32, 3, 56, 56, 1, 1, 1, 0);
  conv(C61O, 32, 56, 56, P(C62_W), P(C62_B), C62O,  3, 3, 56, 56, 1, 1, 1, 0);
  fps_kernel<<<8, 256, 0, stream>>>(C62O, PCGEN, 3136, 2048, 9408, 6144); // raw [8,3136,3] view

  // ---- propagation ----
  gemm(FINAL, P(PR_W1), P(PR_B1), HH,   16384, 64, 3, 3, 64, 64, 1, 1, 0, 0, 0, 0);
  gemm(HH,    P(PR_W2), P(PR_B2), OFFB, 16384,  6, 64, 64, 6,  6, 0, 1, 0, 0, 0, 0);
  {
    long tot = 16384L * 6;
    pred_kernel<<<(int)((tot + 255) / 256), 256, 0, stream>>>(FINAL, OFFB, PRED, tot);
  }
  fps_kernel<<<8, 256, 0, stream>>>(PRED, COMPLETE, 4096, 2048, 12288, 6144);
}
